// Net_43808666419734
// MI455X (gfx1250) — compile-verified, matrix operand order checked by reading, not run on hardware
//
#include <hip/hip_runtime.h>
#include <hip/hip_bf16.h>

// ---------------------------------------------------------------------------
// Fused 2-layer LSTM (H=32) + MLP head for MI455X (gfx1250, wave32, WMMA).
// One wave32 owns a 16-row batch tile and runs the whole sequence with h/c
// register-resident. Gate GEMMs are V_WMMA_F32_16X16X32_F16 (K=32 == H).
// Activations use v_exp_f32 + v_rcp_f32 (no IEEE divide expansions).
// ---------------------------------------------------------------------------

typedef __attribute__((ext_vector_type(16))) _Float16 v16h;
typedef __attribute__((ext_vector_type(8)))  float    v8f;

#define HID    32
#define GATES  128   // 4*H  (i,f,g,o)
#define NT     8     // GATES / 16 N-tiles
#define MROWS  16    // batch rows per wave
#define WAVES  1     // waves per block (1 wave -> 1 WGP per tile, max VGPRs)
#define BATCH  4096
#define SEQ    512

__device__ __forceinline__ float fast_rcp(float x) {
  return __builtin_amdgcn_rcpf(x);            // v_rcp_f32 (trans, co-executes)
}
__device__ __forceinline__ float fast_sig(float x) {
  return fast_rcp(1.0f + __expf(-x));         // v_exp_f32 + v_rcp_f32
}
__device__ __forceinline__ float fast_tanh(float x) {
  // tanh(x) = 2/(1+exp(-2x)) - 1 ; exp->inf => rcp->0 => -1, exp->0 => +1
  return fmaf(2.0f, fast_rcp(1.0f + __expf(-2.0f * x)), -1.0f);
}

// A/B fragment K-mapping for 16-bit 16x16x32 WMMA (wave32):
// lanes 0-15: e0..7 -> K=e,   e8..15 -> K=e+8  (K 0..7, 16..23)
// lanes16-31: e0..7 -> K=e+8, e8..15 -> K=e+16 (K 8..15, 24..31)
__device__ __forceinline__ int kmap_base(int lane) { return (lane & 16) ? 8 : 0; }

// Gather a 16-element f16 fragment for one lane from a 32-wide f16 row.
// Two contiguous 8-half runs -> compiler emits two ds_load_b128.
__device__ __forceinline__ v16h gather16_row(const _Float16* __restrict__ row, int lane) {
  const int kb = kmap_base(lane);
  v16h v;
#pragma unroll
  for (int e = 0; e < 8; ++e) {
    v[e]     = row[kb + e];
    v[e + 8] = row[kb + 16 + e];
  }
  return v;
}

// ---------------------------------------------------------------------------
// Pre-pass: transpose x [B,S] -> xT [S,B] so the recurrence reads coalesced.
// ---------------------------------------------------------------------------
__global__ void transpose_x_kernel(const float* __restrict__ x, float* __restrict__ xT) {
  __shared__ float tile[32][33];
  const int st = blockIdx.x * 32;   // seq tile
  const int bt = blockIdx.y * 32;   // batch tile
  const int tx = threadIdx.x, ty = threadIdx.y;
#pragma unroll
  for (int i = ty; i < 32; i += 8)
    tile[i][tx] = x[(size_t)(bt + i) * SEQ + st + tx];
  __syncthreads();
#pragma unroll
  for (int i = ty; i < 32; i += 8)
    xT[(size_t)(st + i) * BATCH + bt + tx] = tile[tx][i];
}

// ---------------------------------------------------------------------------
// Main fused kernel.
// ---------------------------------------------------------------------------
__global__ __launch_bounds__(WAVES * 32)
void lstm2_wmma_kernel(
    const float* __restrict__ x, int x_is_T,
    const float* __restrict__ w_ih0,                              // [128,1]
    const float* __restrict__ w_hh0,                              // [128,32]
    const float* __restrict__ b_ih0, const float* __restrict__ b_hh0,
    const float* __restrict__ w_ih1,                              // [128,32]
    const float* __restrict__ w_hh1,                              // [128,32]
    const float* __restrict__ b_ih1, const float* __restrict__ b_hh1,
    const float* __restrict__ fc1_w,                              // [16,32]
    const float* __restrict__ fc1_b,                              // [16]
    const float* __restrict__ fc2_w,                              // [16]
    const float* __restrict__ fc2_b,                              // [1]
    float* __restrict__ out)                                      // [B]
{
  __shared__ _Float16 lds_w[GATES * HID];                 // weight repack staging
  __shared__ _Float16 lds_h[WAVES][2][MROWS * HID];       // per-wave h D->A bounce
  __shared__ float    lds_x[WAVES][MROWS];                // x_t broadcast
  __shared__ float    lds_fc[WAVES][MROWS * 16];          // fc1 result bounce

  const int lane = threadIdx.x & 31;
  const int wv   = threadIdx.x >> 5;
  const int b0   = (blockIdx.x * WAVES + wv) * MROWS;     // batch tile start
  const int col  = lane & 15;                             // D-layout column
  const int rofs = (lane & 16) ? 8 : 0;                   // D-layout row offset

  // ---- repack recurrent weights into register-resident B fragments --------
  v16h bW0[NT], bWi1[NT], bWh1[NT];

  for (int i = threadIdx.x; i < GATES * HID; i += blockDim.x)
    lds_w[i] = (_Float16)w_hh0[i];
  __syncthreads();
#pragma unroll
  for (int n = 0; n < NT; ++n) bW0[n] = gather16_row(&lds_w[(n * 16 + col) * HID], lane);
  __syncthreads();

  for (int i = threadIdx.x; i < GATES * HID; i += blockDim.x)
    lds_w[i] = (_Float16)w_ih1[i];
  __syncthreads();
#pragma unroll
  for (int n = 0; n < NT; ++n) bWi1[n] = gather16_row(&lds_w[(n * 16 + col) * HID], lane);
  __syncthreads();

  for (int i = threadIdx.x; i < GATES * HID; i += blockDim.x)
    lds_w[i] = (_Float16)w_hh1[i];
  __syncthreads();
#pragma unroll
  for (int n = 0; n < NT; ++n) bWh1[n] = gather16_row(&lds_w[(n * 16 + col) * HID], lane);

  // per-lane fused biases and the rank-1 input weight of layer 1
  float bias1[NT], bias2[NT], wx[NT];
#pragma unroll
  for (int n = 0; n < NT; ++n) {
    const int g = n * 16 + col;
    bias1[n] = b_ih0[g] + b_hh0[g];
    bias2[n] = b_ih1[g] + b_hh1[g];
    wx[n]    = w_ih0[g];
  }

  // ---- state --------------------------------------------------------------
  v16h a1, a2;                                  // h1 / h2 A-fragments
#pragma unroll
  for (int e = 0; e < 16; ++e) { a1[e] = (_Float16)0.f; a2[e] = (_Float16)0.f; }
  v8f c1[2], c2[2];                             // cell state, D-layout (2 N-tiles of H)
#pragma unroll
  for (int j = 0; j < 2; ++j)
#pragma unroll
    for (int r = 0; r < 8; ++r) { c1[j][r] = 0.f; c2[j][r] = 0.f; }

  _Float16* h1s = &lds_h[wv][0][0];
  _Float16* h2s = &lds_h[wv][1][0];

  // ---- sequential recurrence (parallel only over batch tiles) -------------
  for (int t = 0; t < SEQ; ++t) {
    if (lane < MROWS) {
      float xv = x_is_T ? x[(size_t)t * BATCH + b0 + lane]
                        : x[(size_t)(b0 + lane) * SEQ + t];
      lds_x[wv][lane] = xv;
      if (x_is_T && (t + 8 < SEQ))
        __builtin_prefetch(&x[(size_t)(t + 8) * BATCH + b0 + lane], 0, 0);
    }
    __builtin_amdgcn_wave_barrier();

    float xr[8];
#pragma unroll
    for (int r = 0; r < 8; ++r) xr[r] = lds_x[wv][rofs + r];

    // ---- layer 1: gates = (b_ih+b_hh) + x_t*w_ih0 + h1 @ w_hh0^T  ---------
    v8f acc[NT];
#pragma unroll
    for (int n = 0; n < NT; ++n) {
#pragma unroll
      for (int r = 0; r < 8; ++r) acc[n][r] = fmaf(xr[r], wx[n], bias1[n]);
      acc[n] = __builtin_amdgcn_wmma_f32_16x16x32_f16(
          false, a1, false, bW0[n], (short)0, acc[n], false, false);
    }
    // activations + state update; write h1 rows to LDS for the A-repack
#pragma unroll
    for (int j = 0; j < 2; ++j) {
#pragma unroll
      for (int r = 0; r < 8; ++r) {
        float iv = fast_sig(acc[0 + j][r]);
        float fv = fast_sig(acc[2 + j][r]);
        float gv = fast_tanh(acc[4 + j][r]);
        float ov = fast_sig(acc[6 + j][r]);
        float cc = fmaf(fv, c1[j][r], iv * gv);
        c1[j][r] = cc;
        float hh = ov * fast_tanh(cc);
        h1s[(rofs + r) * HID + j * 16 + col] = (_Float16)hh;
      }
    }
    __builtin_amdgcn_wave_barrier();
    a1 = gather16_row(&h1s[col * HID], lane);   // D-layout -> A-layout

    // ---- layer 2: gates = bias + h1 @ w_ih1^T + h2 @ w_hh1^T --------------
#pragma unroll
    for (int n = 0; n < NT; ++n) {
#pragma unroll
      for (int r = 0; r < 8; ++r) acc[n][r] = bias2[n];
      acc[n] = __builtin_amdgcn_wmma_f32_16x16x32_f16(
          false, a1, false, bWi1[n], (short)0, acc[n], false, false);
      acc[n] = __builtin_amdgcn_wmma_f32_16x16x32_f16(
          false, a2, false, bWh1[n], (short)0, acc[n], false, false);
    }
#pragma unroll
    for (int j = 0; j < 2; ++j) {
#pragma unroll
      for (int r = 0; r < 8; ++r) {
        float iv = fast_sig(acc[0 + j][r]);
        float fv = fast_sig(acc[2 + j][r]);
        float gv = fast_tanh(acc[4 + j][r]);
        float ov = fast_sig(acc[6 + j][r]);
        float cc = fmaf(fv, c2[j][r], iv * gv);
        c2[j][r] = cc;
        float hh = ov * fast_tanh(cc);
        h2s[(rofs + r) * HID + j * 16 + col] = (_Float16)hh;
      }
    }
    __builtin_amdgcn_wave_barrier();
    a2 = gather16_row(&h2s[col * HID], lane);
  }

  // ---- MLP head: leaky_relu(h2 @ fc1_w^T + fc1_b) @ fc2_w^T + fc2_b -------
  v16h bF;
  {
    const float* row = fc1_w + col * HID;       // B[k][n] = fc1_w[n][k]
    const int kb = kmap_base(lane);
#pragma unroll
    for (int e = 0; e < 8; ++e) {
      bF[e]     = (_Float16)row[kb + e];
      bF[e + 8] = (_Float16)row[kb + 16 + e];
    }
  }
  v8f accf;
  {
    float fb = fc1_b[col];
#pragma unroll
    for (int r = 0; r < 8; ++r) accf[r] = fb;
  }
  accf = __builtin_amdgcn_wmma_f32_16x16x32_f16(
      false, a2, false, bF, (short)0, accf, false, false);
#pragma unroll
  for (int r = 0; r < 8; ++r) {
    float v = accf[r];
    v = (v >= 0.f) ? v : 0.2f * v;              // leaky_relu(0.2)
    lds_fc[wv][(rofs + r) * 16 + col] = v;
  }
  __builtin_amdgcn_wave_barrier();
  if (lane < MROWS) {
    float s = fc2_b[0];
#pragma unroll
    for (int j = 0; j < 16; ++j) s = fmaf(lds_fc[wv][lane * 16 + j], fc2_w[j], s);
    out[b0 + lane] = s;
  }
}

// ---------------------------------------------------------------------------
extern "C" void kernel_launch(void* const* d_in, const int* in_sizes, int n_in,
                              void* d_out, int out_size, void* d_ws, size_t ws_size,
                              hipStream_t stream) {
  const float* x     = (const float*)d_in[0];
  const float* w_ih0 = (const float*)d_in[1];
  const float* w_hh0 = (const float*)d_in[2];
  const float* b_ih0 = (const float*)d_in[3];
  const float* b_hh0 = (const float*)d_in[4];
  const float* w_ih1 = (const float*)d_in[5];
  const float* w_hh1 = (const float*)d_in[6];
  const float* b_ih1 = (const float*)d_in[7];
  const float* b_hh1 = (const float*)d_in[8];
  const float* fc1_w = (const float*)d_in[9];
  const float* fc1_b = (const float*)d_in[10];
  const float* fc2_w = (const float*)d_in[11];
  const float* fc2_b = (const float*)d_in[12];
  float* out = (float*)d_out;

  const size_t xT_bytes = (size_t)BATCH * SEQ * sizeof(float);
  const int useT = (d_ws != nullptr && ws_size >= xT_bytes) ? 1 : 0;
  const float* xin = x;
  if (useT) {
    float* xT = (float*)d_ws;
    transpose_x_kernel<<<dim3(SEQ / 32, BATCH / 32), dim3(32, 8), 0, stream>>>(x, xT);
    xin = xT;
  }

  lstm2_wmma_kernel<<<BATCH / (WAVES * MROWS), WAVES * 32, 0, stream>>>(
      xin, useT, w_ih0, w_hh0, b_ih0, b_hh0,
      w_ih1, w_hh1, b_ih1, b_hh1,
      fc1_w, fc1_b, fc2_w, fc2_b, out);
}
